// CriticSwapGNN_4604204941419
// MI455X (gfx1250) — matrix operand order repaired and checked by reading, not compiled, not run on hardware
//
#include <hip/hip_runtime.h>
#include <hip/hip_bf16.h>
#include <math.h>

#define NN  50000
#define NNP 50048              // NN padded to 64-row GEMM blocks (782 * 64)
#define EE  800000
#define NLL 15
#define NGG 8

typedef __attribute__((ext_vector_type(16))) _Float16 v16h;
typedef __attribute__((ext_vector_type(8)))  _Float16 v8h;
typedef __attribute__((ext_vector_type(8)))  float    v8f;

static __device__ __forceinline__ void atomicMaxFloat(float* addr, float val) {
    if (val >= 0.0f) atomicMax((int*)addr, __float_as_int(val));
    else             atomicMin((unsigned int*)addr, __float_as_uint(val));
}

// ---------------------------------------------------------------------------
// Weight pack: W[K x Nc] f32 row-major -> f16 fragments in B-layout order.
// out[((kt*(Nc/16)+ct)*32 + lane)*16 + idx] = W[kt*32 + half*16 + idx][ct*16+l15]
// Kp = K padded up to multiple of 32 (zero fill).
// ---------------------------------------------------------------------------
__global__ void k_pack_w(const float* __restrict__ W, int K, int Nc,
                         _Float16* __restrict__ out, int Kp)
{
    int i = blockIdx.x * blockDim.x + threadIdx.x;
    if (i >= Kp * Nc) return;
    int idx  = i & 15;
    int lane = (i >> 4) & 31;
    int tile = i >> 9;
    int nct  = Nc >> 4;
    int ct   = tile % nct;
    int kt   = tile / nct;
    int half = lane >> 4, l15 = lane & 15;
    int k = kt * 32 + half * 16 + idx;          // K pairs are contiguous
    int n = ct * 16 + l15;
    float v = (k < K) ? W[(size_t)k * Nc + n] : 0.0f;
    out[i] = (_Float16)v;
}

// ---------------------------------------------------------------------------
// WMMA GEMM: Y[NNP x NT*16] = act(Xh[NNP x KT] @ Wpacked + bias)
// All shapes/strides/flags compile-time: pure load+wmma straight-line code,
// epilogue uses one hoisted bias load and immediate-offset stores.
// ---------------------------------------------------------------------------
template<int KT, int NT, int LDX, int LDY, bool HASBIAS, bool RELU, typename TOUT>
__global__ void k_gemm_wmma(const _Float16* __restrict__ Xh,
                            const _Float16* __restrict__ Bp,
                            const float* __restrict__ bias,
                            TOUT* __restrict__ Y)
{
    constexpr int NTILE = 4;                      // 64 rows per block
    const int wave = threadIdx.x >> 5;
    const int lane = threadIdx.x & 31;
    const int half = lane >> 4, l15 = lane & 15;
    const int row0 = blockIdx.x * (16 * NTILE);
    const v16h* __restrict__ Bf = (const v16h*)Bp;

    v8f acc[NTILE] = {};

    #pragma unroll
    for (int kt = 0; kt < KT / 32; ++kt) {
        v16h b = Bf[(size_t)(kt * NT + wave) * 32 + lane];
        #pragma unroll
        for (int t = 0; t < NTILE; ++t) {
            const _Float16* xr = Xh + (size_t)(row0 + t * 16 + l15) * LDX
                               + kt * 32 + half * 16;
            v8h a0 = *(const v8h*)(xr);
            v8h a1 = *(const v8h*)(xr + 8);
            v16h a = __builtin_shufflevector(a0, a1,
                      0,1,2,3,4,5,6,7,8,9,10,11,12,13,14,15);
            acc[t] = __builtin_amdgcn_wmma_f32_16x16x32_f16(
                         false, a, false, b, (short)0, acc[t], false, false);
        }
    }

    const int c = wave * 16 + l15;
    const float bv = HASBIAS ? bias[c] : 0.0f;    // hoisted, one load per lane
    #pragma unroll
    for (int t = 0; t < NTILE; ++t) {
        TOUT* yr = Y + (size_t)(row0 + t * 16 + 8 * half) * LDY + c;
        #pragma unroll
        for (int v = 0; v < 8; ++v) {
            float val = acc[t][v] + bv;
            if (RELU) val = fmaxf(val, 0.0f);
            yr[(size_t)v * LDY] = (TOUT)val;
        }
    }
}

// ---------------------------------------------------------------------------
// small helpers
// ---------------------------------------------------------------------------
__global__ void k_init_scalars(float* scal) {
    int i = threadIdx.x;
    if (i < 32) scal[i] = 0.0f;
}

__global__ void k_req_stats(const float* __restrict__ req, int cnt, float* scal) {
    int i = blockIdx.x * blockDim.x + threadIdx.x;
    float v  = (i < cnt) ? req[NLL + i] : 0.0f;
    float s = v, s2 = v * v;
    #pragma unroll
    for (int off = 16; off > 0; off >>= 1) {
        s  += __shfl_down(s,  off);
        s2 += __shfl_down(s2, off);
    }
    if ((threadIdx.x & 31) == 0) {
        atomicAdd(&scal[0], s);
        atomicAdd(&scal[1], s2);
    }
}

// Build layer-0 node features as f16 with ld = 32 (cols 18..31 zero).
__global__ void k_build_x0(const int* __restrict__ type_ids,
                           const float* __restrict__ update_step,
                           const float* __restrict__ requests,
                           const float* __restrict__ emb,
                           const float* __restrict__ scal,
                           _Float16* __restrict__ x0h, int n)
{
    int i = blockIdx.x * blockDim.x + threadIdx.x;
    if (i >= n) return;
    float cnt  = (float)(n - NLL);
    float mean = scal[0] / cnt;
    float var  = (scal[1] - scal[0] * scal[0] / cnt) / (cnt - 1.0f);
    float stdv = sqrtf(fmaxf(var, 0.0f));
    const float* e = emb + type_ids[i] * 16;
    _Float16* o = x0h + (size_t)i * 32;
    #pragma unroll
    for (int j = 0; j < 16; ++j) o[j] = (_Float16)e[j];
    float r = requests[i];
    o[16] = (_Float16)((i < NLL) ? r : (r - mean) / (stdv + 1e-6f));
    o[17] = (_Float16)update_step[i];
    #pragma unroll
    for (int j = 18; j < 32; ++j) o[j] = (_Float16)0.0f;
}

__global__ void k_we_dot(const float* __restrict__ We,
                         const float* __restrict__ ae, float* scal)
{
    int h = threadIdx.x;
    if (h >= 4) return;
    float s = 0.0f;
    #pragma unroll
    for (int c = 0; c < 32; ++c) s += We[h * 32 + c] * ae[h * 32 + c];
    scal[2 + h] = s;
}

__global__ void k_node_attn(const float* __restrict__ xp,
                            const float* __restrict__ a_s,
                            const float* __restrict__ a_d,
                            float* __restrict__ sa, float* __restrict__ da, int n)
{
    int i = blockIdx.x * blockDim.x + threadIdx.x;
    if (i >= n * 4) return;
    int h = i & 3, node = i >> 2;
    const float* x   = xp  + (size_t)node * 128 + h * 32;
    const float* as_ = a_s + h * 32;
    const float* ad_ = a_d + h * 32;
    float s = 0.0f, d = 0.0f;
    #pragma unroll
    for (int c = 0; c < 32; ++c) { float v = x[c]; s += v * as_[c]; d += v * ad_[c]; }
    sa[i] = s; da[i] = d;
}

__global__ void k_init_head(float* __restrict__ amax, float* __restrict__ denom, int n4) {
    int i = blockIdx.x * blockDim.x + threadIdx.x;
    if (i >= n4) return;
    amax[i]  = __int_as_float(0xff800000);  // -inf
    denom[i] = 0.0f;
}

__global__ void k_zero(float* __restrict__ p, long long n) {
    long long i = (long long)blockIdx.x * blockDim.x + threadIdx.x;
    if (i < n) p[i] = 0.0f;
}

__global__ void k_edge_alpha(const int* __restrict__ src, const int* __restrict__ dst,
                             const float* __restrict__ lat,
                             const float* __restrict__ sa, const float* __restrict__ da,
                             const float* __restrict__ scal,
                             float* __restrict__ alpha, float* __restrict__ amax)
{
    int i = blockIdx.x * blockDim.x + threadIdx.x;
    if (i >= EE * 4) return;
    int h = i & 3, e = i >> 2;
    int s = src[e], d = dst[e];
    float raw = sa[s * 4 + h] + da[d * 4 + h] + lat[e] * scal[2 + h];
    raw = (raw > 0.0f) ? raw : 0.2f * raw;   // leaky_relu 0.2
    alpha[i] = raw;
    atomicMaxFloat(&amax[d * 4 + h], raw);
}

__global__ void k_edge_exp(const int* __restrict__ dst,
                           const float* __restrict__ amax,
                           float* __restrict__ alpha, float* __restrict__ denom)
{
    int i = blockIdx.x * blockDim.x + threadIdx.x;
    if (i >= EE * 4) return;
    int h = i & 3, e = i >> 2;
    int d = dst[e];
    float v = expf(alpha[i] - amax[d * 4 + h]);
    alpha[i] = v;
    atomicAdd(&denom[d * 4 + h], v);
}

__global__ void k_edge_aggr(const int* __restrict__ src, const int* __restrict__ dst,
                            const float* __restrict__ xp,
                            const float* __restrict__ alpha,
                            const float* __restrict__ denom,
                            float* __restrict__ out)
{
    long long i = (long long)blockIdx.x * blockDim.x + threadIdx.x;
    if (i >= (long long)EE * 128) return;
    int j = (int)(i & 127);
    int e = (int)(i >> 7);
    int h = j >> 5;
    int s = src[e], d = dst[e];
    float a = alpha[e * 4 + h] / (denom[d * 4 + h] + 1e-16f);
    atomicAdd(&out[(size_t)d * 128 + j], xp[(size_t)s * 128 + j] * a);
}

// bias + optional relu; f32 accumulated features -> f16 feature buffer
__global__ void k_bias_act(const float* __restrict__ fin, const float* __restrict__ b,
                           _Float16* __restrict__ fout, int relu, long long n)
{
    long long i = (long long)blockIdx.x * blockDim.x + threadIdx.x;
    if (i >= n) return;
    int j = (int)(i & 127);
    float v = fin[i] + b[j];
    if (relu) v = fmaxf(v, 0.0f);
    fout[i] = (_Float16)v;
}

__global__ void k_final_dot(const float* __restrict__ h2,
                            const float* __restrict__ C3w, const float* __restrict__ C3b,
                            const int* __restrict__ batch,
                            float* __restrict__ gsum, float* __restrict__ gcnt, int n)
{
    int wid  = (int)(((long long)blockIdx.x * blockDim.x + threadIdx.x) >> 5);
    int lane = threadIdx.x & 31;
    if (wid >= n) return;
    const float* r = h2 + (size_t)wid * 256;
    float s = 0.0f;
    for (int j = lane; j < 256; j += 32) s += r[j] * C3w[j];
    #pragma unroll
    for (int off = 16; off > 0; off >>= 1) s += __shfl_down(s, off);
    if (lane == 0) {
        float v = fmaxf(s + C3b[0], 0.0f);
        int g = batch[wid];
        atomicAdd(&gsum[g], v);
        atomicAdd(&gcnt[g], 1.0f);
    }
}

__global__ void k_group_out(const float* __restrict__ gsum,
                            const float* __restrict__ gcnt,
                            float* __restrict__ out)
{
    int g = threadIdx.x;
    if (g < NGG) out[g] = gsum[g] / fmaxf(gcnt[g], 1.0f);
}

// ---------------------------------------------------------------------------
static inline int cdiv(long long a, int b) { return (int)((a + b - 1) / b); }

extern "C" void kernel_launch(void* const* d_in, const int* in_sizes, int n_in,
                              void* d_out, int out_size, void* d_ws, size_t ws_size,
                              hipStream_t stream)
{
    (void)in_sizes; (void)n_in; (void)out_size; (void)ws_size;

    const int*   type_ids    = (const int*)  d_in[0];
    const float* update_step = (const float*)d_in[1];
    const float* requests    = (const float*)d_in[2];
    const int*   edge_index  = (const int*)  d_in[3];
    const float* latency     = (const float*)d_in[4];
    const int*   batch       = (const int*)  d_in[5];
    const float* emb         = (const float*)d_in[6];
    const float* W0  = (const float*)d_in[7];
    const float* We0 = (const float*)d_in[8];
    const float* as0 = (const float*)d_in[9];
    const float* ad0 = (const float*)d_in[10];
    const float* ae0 = (const float*)d_in[11];
    const float* b0  = (const float*)d_in[12];
    const float* Wh  = (const float*)d_in[13];
    const float* Weh = (const float*)d_in[14];
    const float* ash = (const float*)d_in[15];
    const float* adh = (const float*)d_in[16];
    const float* aeh = (const float*)d_in[17];
    const float* bh  = (const float*)d_in[18];
    const float* Wf  = (const float*)d_in[19];
    const float* Wef = (const float*)d_in[20];
    const float* asf = (const float*)d_in[21];
    const float* adf = (const float*)d_in[22];
    const float* aef = (const float*)d_in[23];
    const float* bf  = (const float*)d_in[24];
    const float* C1w = (const float*)d_in[25];
    const float* C1b = (const float*)d_in[26];
    const float* C2w = (const float*)d_in[27];
    const float* C2b = (const float*)d_in[28];
    const float* C3w = (const float*)d_in[29];
    const float* C3b = (const float*)d_in[30];

    const int* src = edge_index;
    const int* dst = edge_index + EE;
    float* out = (float*)d_out;

    // workspace carve-up (256B aligned slabs); GEMM buffers padded to NNP rows
    size_t off = 0;
    auto allocB = [&](size_t bytes) {
        void* p = (char*)d_ws + off;
        off += (bytes + 255) & ~(size_t)255;
        return p;
    };
    float* scal  = (float*)allocB(32 * 4);
    float* xp    = (float*)allocB((size_t)NNP * 128 * 4);
    float* facc  = (float*)allocB((size_t)NN  * 128 * 4);    // f32 aggregation accum
    float* h2    = (float*)allocB((size_t)NNP * 256 * 4);
    float* sa    = (float*)allocB((size_t)NN * 4 * 4);
    float* da    = (float*)allocB((size_t)NN * 4 * 4);
    float* amax  = (float*)allocB((size_t)NN * 4 * 4);
    float* denom = (float*)allocB((size_t)NN * 4 * 4);
    float* alpha = (float*)allocB((size_t)EE * 4 * 4);
    _Float16* x0h  = (_Float16*)allocB((size_t)NNP * 32 * 2);
    _Float16* ftA  = (_Float16*)allocB((size_t)NNP * 128 * 2);
    _Float16* ftB  = (_Float16*)allocB((size_t)NNP * 128 * 2);
    _Float16* h1h  = (_Float16*)allocB((size_t)NNP * 256 * 2);
    _Float16* W0p  = (_Float16*)allocB((size_t)32  * 128 * 2);
    _Float16* Whp0 = (_Float16*)allocB((size_t)128 * 128 * 2);
    _Float16* Whp1 = (_Float16*)allocB((size_t)128 * 128 * 2);
    _Float16* Wfp  = (_Float16*)allocB((size_t)128 * 128 * 2);
    _Float16* C1p  = (_Float16*)allocB((size_t)128 * 256 * 2);
    _Float16* C2p  = (_Float16*)allocB((size_t)256 * 256 * 2);
    float* gsum = scal + 6;
    float* gcnt = scal + 14;

    // ---- weight packing (f32 -> f16 B-fragment layout) -------------------
    k_pack_w<<<cdiv(32  * 128, 256), 256, 0, stream>>>(W0,             18,  128, W0p,  32);
    k_pack_w<<<cdiv(128 * 128, 256), 256, 0, stream>>>(Wh,             128, 128, Whp0, 128);
    k_pack_w<<<cdiv(128 * 128, 256), 256, 0, stream>>>(Wh + 128 * 128, 128, 128, Whp1, 128);
    k_pack_w<<<cdiv(128 * 128, 256), 256, 0, stream>>>(Wf,             128, 128, Wfp,  128);
    k_pack_w<<<cdiv(128 * 256, 256), 256, 0, stream>>>(C1w,            128, 256, C1p,  128);
    k_pack_w<<<cdiv(256 * 256, 256), 256, 0, stream>>>(C2w,            256, 256, C2p,  256);

    // ---- input feature build --------------------------------------------
    k_init_scalars<<<1, 32, 0, stream>>>(scal);
    k_req_stats<<<cdiv(NN - NLL, 256), 256, 0, stream>>>(requests, NN - NLL, scal);
    k_build_x0<<<cdiv(NN, 256), 256, 0, stream>>>(type_ids, update_step, requests,
                                                  emb, scal, x0h, NN);

    const int gGemm = NNP / 64;   // 782 blocks, zero ragged edges

    // ---- GAT attention/aggregation stage (xp already computed) -----------
    auto gat_tail = [&](const float* We, const float* a_s, const float* a_d,
                        const float* a_e, const float* b, _Float16* fout, int relu)
    {
        k_we_dot<<<1, 4, 0, stream>>>(We, a_e, scal);
        k_node_attn<<<cdiv((long long)NN * 4, 256), 256, 0, stream>>>(xp, a_s, a_d, sa, da, NN);
        k_init_head<<<cdiv((long long)NN * 4, 256), 256, 0, stream>>>(amax, denom, NN * 4);
        k_zero<<<cdiv((long long)NN * 128, 256), 256, 0, stream>>>(facc, (long long)NN * 128);
        k_edge_alpha<<<cdiv((long long)EE * 4, 256), 256, 0, stream>>>(src, dst, latency,
                                                                       sa, da, scal, alpha, amax);
        k_edge_exp<<<cdiv((long long)EE * 4, 256), 256, 0, stream>>>(dst, amax, alpha, denom);
        k_edge_aggr<<<cdiv((long long)EE * 128, 256), 256, 0, stream>>>(src, dst, xp,
                                                                        alpha, denom, facc);
        k_bias_act<<<cdiv((long long)NN * 128, 256), 256, 0, stream>>>(facc, b, fout, relu,
                                                                       (long long)NN * 128);
    };

    // layer 0 (din 18 padded to 32)
    k_gemm_wmma<32, 8, 32, 128, false, false, float>
        <<<gGemm, 256, 0, stream>>>(x0h, W0p, nullptr, xp);
    gat_tail(We0, as0, ad0, ae0, b0, ftA, 1);

    // hidden layer 1
    k_gemm_wmma<128, 8, 128, 128, false, false, float>
        <<<gGemm, 256, 0, stream>>>(ftA, Whp0, nullptr, xp);
    gat_tail(Weh, ash, adh, aeh, bh, ftB, 1);

    // hidden layer 2
    k_gemm_wmma<128, 8, 128, 128, false, false, float>
        <<<gGemm, 256, 0, stream>>>(ftB, Whp1, nullptr, xp);
    gat_tail(Weh + 128, ash + 128, adh + 128, aeh + 128, bh + 128, ftA, 1);

    // final GAT layer (no relu)
    k_gemm_wmma<128, 8, 128, 128, false, false, float>
        <<<gGemm, 256, 0, stream>>>(ftA, Wfp, nullptr, xp);
    gat_tail(Wef, asf, adf, aef, bf, ftB, 0);

    // ---- critic MLP (WMMA) ----------------------------------------------
    k_gemm_wmma<128, 16, 128, 256, true, true, _Float16>
        <<<gGemm, 512, 0, stream>>>(ftB, C1p, C1b, h1h);
    k_gemm_wmma<256, 16, 256, 256, true, true, float>
        <<<gGemm, 512, 0, stream>>>(h1h, C2p, C2b, h2);
    k_final_dot<<<cdiv(NN, 8), 256, 0, stream>>>(h2, C3w, C3b, batch, gsum, gcnt, NN);
    k_group_out<<<1, NGG, 0, stream>>>(gsum, gcnt, out);
}
